// DecagonModel_9242769622248
// MI455X (gfx1250) — matrix operand order repaired and checked by reading, not compiled, not run on hardware
//
#include <hip/hip_runtime.h>
#include <hip/hip_bf16.h>

// ---------------------------------------------------------------------------
// Decagon hetero-GCN forward for MI455X (gfx1250, wave32, WMMA).
// Dense X@W via v_wmma_f32_16x16x32_bf16 (bf16 A/B, f32 accumulate).
// Sparse A@Y via wave-per-edge gather + f32 global atomics (L2-resident).
// ---------------------------------------------------------------------------

typedef __attribute__((ext_vector_type(16))) __bf16 v16bf;
typedef __attribute__((ext_vector_type(2)))  __bf16 v2bf;
typedef __attribute__((ext_vector_type(8)))  float  v8f;

__device__ inline unsigned int f32_to_bf16_rne(float f) {
    unsigned int x = __float_as_uint(f);
    return (x + 0x7FFFu + ((x >> 16) & 1u)) >> 16;   // round-to-nearest-even
}

__device__ inline unsigned int pack_bf16x2(float lo, float hi) {
#if __has_builtin(__builtin_amdgcn_cvt_pk_bf16_f32)
    union { v2bf h; unsigned int u; } cv;
    cv.h = __builtin_amdgcn_cvt_pk_bf16_f32(lo, hi);   // v_cvt_pk_bf16_f32
    return cv.u;
#else
    return f32_to_bf16_rne(lo) | (f32_to_bf16_rne(hi) << 16);
#endif
}

union Frag16 {                 // 16 bf16 = 8 dwords = two 16-byte chunks
    uint4        q[2];
    unsigned int u32[8];
    v16bf        v;
};

// ---------------------------------------------------------------------------
// Y[N x 64] = X[N x K] @ W[K x 64]   (K = 128 or 64; X has row stride ldx)
// Block: 256 threads = 8 waves; each wave computes 16 rows x 64 cols.
// W is staged in LDS transposed + bf16-pair-packed: Wlds[col][k/2] (uint),
// row stride K/2+4 dwords -> each B fragment is two ds_load_b128.
// ---------------------------------------------------------------------------
#define GEMM_TILE_M 128
#define WLDS_MAX    (64 * (128 / 2 + 4))     // 17408 B

__global__ __launch_bounds__(256)
void gemm_bf16_wmma(const float* __restrict__ X, int ldx, int K,
                    const float* __restrict__ W,     // K x 64, row-major, f32
                    float* __restrict__ Y, int nRows)
{
    __shared__ unsigned int Wlds[WLDS_MAX];

    const int tid     = threadIdx.x;
    const int kHalf   = K >> 1;              // bf16 pairs per column
    const int strideU = kHalf + 4;           // padded row stride (dwords)

    // Stage W^T into LDS, two bf16 per dword: Wlds[n*strideU + kp] = {W[2kp][n], W[2kp+1][n]}
    for (int idx = tid; idx < 64 * kHalf; idx += 256) {
        const int n  = idx / kHalf;
        const int kp = idx - n * kHalf;
        Wlds[n * strideU + kp] =
            pack_bf16x2(W[(2 * kp) * 64 + n], W[(2 * kp + 1) * 64 + n]);
    }
    __syncthreads();

    const int wave = tid >> 5;               // 0..7
    const int lane = tid & 31;
    const int m16  = lane & 15;
    const int half = lane >> 4;              // 0 or 1
    const int rowBase = blockIdx.x * GEMM_TILE_M + wave * 16;
    int rowLd = rowBase + m16;
    if (rowLd >= nRows) rowLd = nRows - 1;   // clamp OOB loads; stores guarded

    v8f acc[4] = {v8f{}, v8f{}, v8f{}, v8f{}};

    for (int kc = 0; kc < K; kc += 32) {
        // ---- A fragment: 16x32 bf16.
        // Lanes 0-15 hold row m16, K = kc..kc+15;
        // lanes 16-31 hold K = kc+16..kc+31.
        const float* xr = X + (long)rowLd * ldx + kc + half * 16;
        const float4 f0 = *(const float4*)(xr);
        const float4 f1 = *(const float4*)(xr + 4);
        const float4 f2 = *(const float4*)(xr + 8);
        const float4 f3 = *(const float4*)(xr + 12);
        Frag16 a;
        a.u32[0] = pack_bf16x2(f0.x, f0.y);
        a.u32[1] = pack_bf16x2(f0.z, f0.w);
        a.u32[2] = pack_bf16x2(f1.x, f1.y);
        a.u32[3] = pack_bf16x2(f1.z, f1.w);
        a.u32[4] = pack_bf16x2(f2.x, f2.y);
        a.u32[5] = pack_bf16x2(f2.z, f2.w);
        a.u32[6] = pack_bf16x2(f3.x, f3.y);
        a.u32[7] = pack_bf16x2(f3.z, f3.w);

        // ---- 4 column blocks of B (32x16 bf16 each) and WMMA.
        const int kb = kc + half * 16;       // lanes 0-15: K=kc..+15; 16-31: +16
#pragma unroll
        for (int c = 0; c < 4; ++c) {
            const int n = m16 + c * 16;
            const unsigned int* bp = &Wlds[n * strideU + (kb >> 1)];
            Frag16 b;
            b.q[0] = *(const uint4*)(bp);        // ds_load_b128
            b.q[1] = *(const uint4*)(bp + 4);    // ds_load_b128
            acc[c] = __builtin_amdgcn_wmma_f32_16x16x32_bf16(
                false, a.v, false, b.v, (short)0, acc[c], false, false);
        }
    }

    // ---- Store C/D: VGPR r -> row rowBase + r + half*8, col = m16 + c*16.
    if (rowBase + 16 <= nRows) {
        // Fast path (all but the last block): one base pointer per lane,
        // 32 plain global_store_b32 with immediate offsets.
        float* yp = Y + (long)(rowBase + half * 8) * 64 + m16;
#pragma unroll
        for (int c = 0; c < 4; ++c)
#pragma unroll
            for (int r = 0; r < 8; ++r)
                yp[r * 64 + c * 16] = acc[c][r];
    } else if (rowBase < nRows) {
        // Partial tile: per-element guard.
#pragma unroll
        for (int c = 0; c < 4; ++c) {
            const int col = m16 + c * 16;
#pragma unroll
            for (int r = 0; r < 8; ++r) {
                const int row = rowBase + r + half * 8;
                if (row < nRows)
                    Y[(long)row * 64 + col] = acc[c][r];
            }
        }
    }
}

// ---------------------------------------------------------------------------
// out[dst[e]] += ew[e] * Y[src[e]]   (64 features; one wave32 per edge,
// 2 features per lane). out has row stride ldo.
// ---------------------------------------------------------------------------
__global__ __launch_bounds__(256)
void spmm_scatter(const float* __restrict__ Y,
                  const int* __restrict__ src,
                  const int* __restrict__ dst,
                  const float* __restrict__ ew,
                  float* __restrict__ out, int ldo, int nE)
{
    const long gtid = (long)blockIdx.x * blockDim.x + threadIdx.x;
    const long e    = gtid >> 5;
    const int  lane = (int)(gtid & 31);
    if (e >= nE) return;

    const int   s = src[e];
    const int   d = dst[e];
    const float w = ew[e];

    const float* yr = Y + (long)s * 64;
    float v0 = yr[lane]      * w;
    float v1 = yr[lane + 32] * w;

    float* orow = out + (long)d * ldo;
    atomicAdd(orow + lane,      v0);
    atomicAdd(orow + lane + 32, v1);
}

// ---------------------------------------------------------------------------
// Zero / ReLU over an N x 64 region with row stride ldo.
// ---------------------------------------------------------------------------
__global__ __launch_bounds__(256)
void zero_strided(float* __restrict__ p, int ldo, long n)
{
    long i = (long)blockIdx.x * blockDim.x + threadIdx.x;
    if (i < n) p[(i >> 6) * (long)ldo + (i & 63)] = 0.0f;
}

__global__ __launch_bounds__(256)
void relu_strided(float* __restrict__ p, int ldo, long n)
{
    long i = (long)blockIdx.x * blockDim.x + threadIdx.x;
    if (i < n) {
        long off = (i >> 6) * (long)ldo + (i & 63);
        float v = p[off];
        p[off] = v > 0.0f ? v : 0.0f;
    }
}

// ---------------------------------------------------------------------------
extern "C" void kernel_launch(void* const* d_in, const int* in_sizes, int n_in,
                              void* d_out, int out_size, void* d_ws, size_t ws_size,
                              hipStream_t stream)
{
    const float* x0  = (const float*)d_in[0];
    const float* x1  = (const float*)d_in[1];
    const int*   src = (const int*)  d_in[2];
    const int*   dst = (const int*)  d_in[3];
    const float* ew  = (const float*)d_in[4];
    const float* W1  = (const float*)d_in[5];   // [4][128][64]
    const float* Wl  = (const float*)d_in[6];   // [4][4][64][64]
    float* out = (float*)d_out;                 // (2, N, 192)

    const int F = 128, H = 64;
    const int N = in_sizes[0] / F;              // 50000
    const int E = in_sizes[2] / 4;              // 800000

    // d_out layout per type t: rows of 192 = [h1 | e1 | emb]
    float* base0 = out;
    float* base1 = out + (long)N * 192;

    // workspace: e2 (2 x N*64), e3 (2 x N*64), ybuf (N*64)
    float* e2_0 = (float*)d_ws;
    float* e2_1 = e2_0 + (long)N * 64;
    float* e3_0 = e2_1 + (long)N * 64;
    float* e3_1 = e3_0 + (long)N * 64;
    float* ybuf = e3_1 + (long)N * 64;

    const long nNH       = (long)N * 64;
    const int  zrBlocks  = (int)((nNH + 255) / 256);
    const int  gemmBlks  = (N + GEMM_TILE_M - 1) / GEMM_TILE_M;
    const int  scatBlks  = (int)(((long)E * 32 + 255) / 256);

    // One hetero-GCN layer: outs[i] += A_k @ (in[j] @ W_k), k = 0..3
    auto run_layer = [&](const float* in0, int ld0, const float* in1, int ld1, int K,
                         const float* Wbase, long wStride,
                         float* out0, float* out1, int ldo, bool relu)
    {
        zero_strided<<<zrBlocks, 256, 0, stream>>>(out0, ldo, nNH);
        zero_strided<<<zrBlocks, 256, 0, stream>>>(out1, ldo, nNH);
        for (int k = 0; k < 4; ++k) {
            const int i = k / 2, j = k % 2;
            const float* Xin = j ? in1 : in0;
            const int    ldx = j ? ld1 : ld0;
            gemm_bf16_wmma<<<gemmBlks, 256, 0, stream>>>(
                Xin, ldx, K, Wbase + (long)k * wStride, ybuf, N);
            float* o = i ? out1 : out0;
            spmm_scatter<<<scatBlks, 256, 0, stream>>>(
                ybuf, src + (long)k * E, dst + (long)k * E, ew + (long)k * E,
                o, ldo, E);
        }
        if (relu) {
            relu_strided<<<zrBlocks, 256, 0, stream>>>(out0, ldo, nNH);
            relu_strided<<<zrBlocks, 256, 0, stream>>>(out1, ldo, nNH);
        }
    };

    const long wl = (long)H * H;            // 64*64 per relation
    // layer 1: x -> h1 (d_out cols [0,64), stride 192)
    run_layer(x0, F, x1, F, F, W1, (long)F * H,
              base0 + 0,   base1 + 0,   192, true);
    // layer 2: h1 -> e1 (d_out cols [64,128))
    run_layer(base0 + 0,  192, base1 + 0,  192, H, Wl + 0 * 4 * wl, wl,
              base0 + 64,  base1 + 64,  192, true);
    // layer 3: e1 -> e2 (ws)
    run_layer(base0 + 64, 192, base1 + 64, 192, H, Wl + 1 * 4 * wl, wl,
              e2_0, e2_1, 64, true);
    // layer 4: e2 -> e3 (ws)
    run_layer(e2_0, 64, e2_1, 64, H, Wl + 2 * 4 * wl, wl,
              e3_0, e3_1, 64, true);
    // layer 5: e3 -> emb (d_out cols [128,192)), no ReLU
    run_layer(e3_0, 64, e3_1, 64, H, Wl + 3 * 4 * wl, wl,
              base0 + 128, base1 + 128, 192, false);
}
